// GraphTransformerBlock_55173149884913
// MI455X (gfx1250) — matrix operand order repaired and checked by reading, not compiled, not run on hardware
//
#include <hip/hip_runtime.h>
#include <hip/hip_bf16.h>
#include <math.h>

#define NNODES 50000
#define NEDGES 800000
// H=64, NH=4, DK=16

typedef __attribute__((ext_vector_type(16))) __bf16 v16bf;
typedef __attribute__((ext_vector_type(8)))  float  v8f;
typedef __attribute__((ext_vector_type(4)))  unsigned int u32x4;

union FragBF { v16bf v; u32x4 u[2]; };

// ---- A fragment (16x32 bf16 tile) from row-major LDS [16][lda] ----
// ISA layout: lanes 0-15: M=lane, K in [kb, kb+8) and [kb+16, kb+24)
//             lanes 16-31: M=lane-16, K in [kb+8, kb+16) and [kb+24, kb+32)
static __device__ inline v16bf ldsA(const __bf16* a, int lda, int kb, int lane) {
    const int half = lane >> 4, m = lane & 15;
    const __bf16* p = a + m * lda + kb + half * 8;
    FragBF f;
    f.u[0] = *(const u32x4*)(p);        // 8 bf16 = 16B
    f.u[1] = *(const u32x4*)(p + 16);   // next chunk
    return f.v;
}

// ---- B fragment (32x16 bf16 tile) from bf16 weights W[64][ldw] (row-major) ----
// B[k][n] = W[ntile*16+n][k]; lane = column n (+half h selects K [8h,8h+8) and [16+8h,24+8h)),
// so each lane reads two 8-bf16 chunks from one weight row.
static __device__ inline v16bf gblB(const __bf16* w, int ldw, int ntile, int kb, int lane) {
    const int n = lane & 15;
    const int koff = kb + ((lane >> 4) << 3);
    const __bf16* p = w + (ntile * 16 + n) * ldw + koff;
    FragBF f;
    f.u[0] = *(const u32x4*)(p);
    f.u[1] = *(const u32x4*)(p + 16);
    return f.v;
}

static __device__ inline v8f wmma_bf16(v16bf a, v16bf b, v8f c) {
    return __builtin_amdgcn_wmma_f32_16x16x32_bf16(false, a, false, b, (short)0, c, false, false);
}

// ---- one MLP layer: D[16][64] = act(A[16][K] @ W.T + bias) ----
// k OUTER / n-tile INNER with 4 live accumulators so consecutive WMMAs hit
// different D registers (avoids the WMMA->WMMA RAW hazard NOPs, ISA 7.12.1).
template<int K, bool RELU>
static __device__ inline void mlp_layer(const __bf16* A, int lda,
                                        const __bf16* W, const float* bias,
                                        __bf16* outBF, float* outF32, int lane) {
    constexpr int NK = K / 32;
    const int ncol = lane & 15, mbase = (lane >> 4) * 8;
    v8f acc[4];
#pragma unroll
    for (int t = 0; t < 4; ++t) {
        float bv = bias ? bias[t * 16 + ncol] : 0.0f;
#pragma unroll
        for (int r = 0; r < 8; ++r) acc[t][r] = bv;   // C init = bias (per column)
    }
#pragma unroll
    for (int k = 0; k < NK; ++k) {
        v16bf a = ldsA(A, lda, k * 32, lane);
#pragma unroll
        for (int t = 0; t < 4; ++t)
            acc[t] = wmma_bf16(a, gblB(W, K, t, k * 32, lane), acc[t]);
    }
#pragma unroll
    for (int t = 0; t < 4; ++t) {
        const int n = t * 16 + ncol;
#pragma unroll
        for (int r = 0; r < 8; ++r) {
            float v = acc[t][r];
            if (RELU) v = v > 0.0f ? v : 0.0f;
            if (outBF)  outBF[(mbase + r) * 64 + n] = (__bf16)v;
            if (outF32) outF32[(mbase + r) * 64 + n] = v;
        }
    }
}

// ---- 64->64 projection helper (K=64), result kept in 4 accumulators ----
static __device__ inline void proj64(const __bf16* A, const __bf16* W,
                                     v8f acc[4], int lane) {
#pragma unroll
    for (int t = 0; t < 4; ++t)
#pragma unroll
        for (int r = 0; r < 8; ++r) acc[t][r] = 0.0f;
#pragma unroll
    for (int k = 0; k < 2; ++k) {
        v16bf a = ldsA(A, 64, k * 32, lane);
#pragma unroll
        for (int t = 0; t < 4; ++t)
            acc[t] = wmma_bf16(a, gblB(W, 64, t, k * 32, lane), acc[t]);
    }
}

// ---- LayerNorm over 64 channels for 16 rows resident in LDS (one wave) ----
static __device__ inline void layer_norm16(float* hf, const float* g, const float* be,
                                           float* stats, int lane) {
    if (lane < 16) {
        float s = 0.f, ss = 0.f;
        for (int c = 0; c < 64; ++c) { float v = hf[lane * 64 + c]; s += v; ss += v * v; }
        float mu = s * (1.0f / 64.0f);
        float var = ss * (1.0f / 64.0f) - mu * mu;
        stats[lane * 2 + 0] = mu;
        stats[lane * 2 + 1] = rsqrtf(var + 1e-5f);
    }
    __builtin_amdgcn_wave_barrier();   // keep LDS write->read ordering within wave
    for (int i = lane; i < 16 * 64; i += 32) {
        int r = i >> 6, c = i & 63;
        hf[i] = (hf[i] - stats[r * 2]) * stats[r * 2 + 1] * g[c] + be[c];
    }
}

// ---------------- weight fp32 -> bf16 packing ----------------
// wbf layout (bf16 elems): e_w0@0(12288) e_w1@12288 e_w2@16384 wq@20480 wk@24576
//                          wv@28672 wo@32768 n_w0@36864(8192) n_w1@45056 n_w2@49152
__global__ void wconv_kernel(const float* e_w0, const float* e_w1, const float* e_w2,
                             const float* wq, const float* wk, const float* wvp, const float* wo,
                             const float* n_w0, const float* n_w1, const float* n_w2,
                             __bf16* wbf) {
    int i = blockIdx.x * blockDim.x + threadIdx.x;
    float v;
    if      (i < 12288) v = e_w0[i];
    else if (i < 16384) v = e_w1[i - 12288];
    else if (i < 20480) v = e_w2[i - 16384];
    else if (i < 24576) v = wq[i - 20480];
    else if (i < 28672) v = wk[i - 24576];
    else if (i < 32768) v = wvp[i - 28672];
    else if (i < 36864) v = wo[i - 32768];
    else if (i < 45056) v = n_w0[i - 36864];
    else if (i < 49152) v = n_w1[i - 45056];
    else if (i < 53248) v = n_w2[i - 49152];
    else return;
    wbf[i] = (__bf16)v;
}

__global__ void init_kernel(unsigned int* smax_u, float* sumexp, float* msg) {
    int i = blockIdx.x * blockDim.x + threadIdx.x;
    if (i < NNODES * 4) { smax_u[i] = 0u; sumexp[i] = 0.0f; }
    if (i < NNODES * 64) msg[i] = 0.0f;
}

// ---------------- edge kernel: edge MLP + QKV + per-edge scores ----------------
struct EdgeSmem {
    __bf16 inA[16 * 192];   // [x_dst | x_src | edge_attr] bf16
    __bf16 hbf[16 * 64];    // bf16 activations
    float  hf[16 * 64];     // fp32 for LayerNorm
    float  stats[32];
    float  sc[16 * 4];      // scores
};

__global__ __launch_bounds__(128) void edge_kernel(
        const float* __restrict__ x, const int* __restrict__ ei,
        const float* __restrict__ eattr, const __bf16* __restrict__ wbf,
        const float* e_b0, const float* e_b1, const float* e_b2,
        const float* e_g, const float* e_be,
        float* __restrict__ e_out, float* __restrict__ Vbuf, float* __restrict__ scoreg) {
    __shared__ EdgeSmem smem[4];
    const int lane = threadIdx.x & 31;
    const int wv = threadIdx.x >> 5;
    EdgeSmem& S = smem[wv];
    const int e0 = (blockIdx.x * 4 + wv) * 16;
    if (e0 >= NEDGES) return;
    const int* srcI = ei;
    const int* dstI = ei + NEDGES;

    const __bf16* w_e0 = wbf + 0;
    const __bf16* w_e1 = wbf + 12288;
    const __bf16* w_e2 = wbf + 16384;
    const __bf16* w_q  = wbf + 20480;
    const __bf16* w_k  = wbf + 24576;
    const __bf16* w_v  = wbf + 28672;

    // prefetch next tile's edge_attr into cache hierarchy
    if (e0 + 16 < NEDGES) __builtin_prefetch(eattr + (size_t)(e0 + 16) * 64, 0, 1);

    // ---- gather + concat + fp32->bf16 ----
    const int c = lane * 2;
    for (int r = 0; r < 16; ++r) {
        int e = e0 + r;
        int vd = dstI[e], vs = srcI[e];
        float2 a = *(const float2*)(x + (size_t)vd * 64 + c);
        float2 b = *(const float2*)(x + (size_t)vs * 64 + c);
        float2 w = *(const float2*)(eattr + (size_t)e * 64 + c);
        S.inA[r * 192 + c]           = (__bf16)a.x; S.inA[r * 192 + c + 1]       = (__bf16)a.y;
        S.inA[r * 192 + 64 + c]      = (__bf16)b.x; S.inA[r * 192 + 64 + c + 1]  = (__bf16)b.y;
        S.inA[r * 192 + 128 + c]     = (__bf16)w.x; S.inA[r * 192 + 128 + c + 1] = (__bf16)w.y;
    }

    // ---- edge MLP: 192->64 relu, 64->64 relu, 64->64, LayerNorm ----
    mlp_layer<192, true >(S.inA, 192, w_e0, e_b0, S.hbf, nullptr, lane);
    mlp_layer<64,  true >(S.hbf,  64, w_e1, e_b1, S.hbf, nullptr, lane);
    mlp_layer<64,  false>(S.hbf,  64, w_e2, e_b2, nullptr, S.hf, lane);
    layer_norm16(S.hf, e_g, e_be, S.stats, lane);

    // ---- e_out = edge_attr + e_new ; refresh bf16 copy of e_new ----
    for (int i = lane; i < 16 * 64; i += 32) {
        int r = i >> 6, cc = i & 63;
        size_t e = (size_t)(e0 + r);
        float v = S.hf[i];
        e_out[e * 64 + cc] = eattr[e * 64 + cc] + v;
        S.hbf[i] = (__bf16)v;
    }

    const int ncol = lane & 15, mbase = (lane >> 4) * 8;

    // ---- V = e_new @ wv.T -> global scratch ----
    {
        v8f vacc[4];
        proj64(S.hbf, w_v, vacc, lane);
#pragma unroll
        for (int t = 0; t < 4; ++t) {
            int n = t * 16 + ncol;
#pragma unroll
            for (int r = 0; r < 8; ++r)
                Vbuf[(size_t)(e0 + mbase + r) * 64 + n] = vacc[t][r];
        }
    }

    // ---- scores: Q = x_dst@wq.T, K = x_src@wk.T; head h == output n-tile h ----
    // 8 independent accumulator chains (4 heads x {q,k}) interleaved per k-chunk.
    {
        v8f qacc[4], kacc[4];
#pragma unroll
        for (int h = 0; h < 4; ++h)
#pragma unroll
            for (int r = 0; r < 8; ++r) { qacc[h][r] = 0.0f; kacc[h][r] = 0.0f; }
#pragma unroll
        for (int k = 0; k < 2; ++k) {
            v16bf qa = ldsA(S.inA, 192, k * 32, lane);       // x_dst cols
            v16bf ka = ldsA(S.inA, 192, 64 + k * 32, lane);  // x_src cols
#pragma unroll
            for (int h = 0; h < 4; ++h) {
                qacc[h] = wmma_bf16(qa, gblB(w_q, 64, h, k * 32, lane), qacc[h]);
                kacc[h] = wmma_bf16(ka, gblB(w_k, 64, h, k * 32, lane), kacc[h]);
            }
        }
#pragma unroll
        for (int h = 0; h < 4; ++h) {
            float p[8];
#pragma unroll
            for (int r = 0; r < 8; ++r) p[r] = qacc[h][r] * kacc[h][r];
            // sum over the 16 columns of this head (lanes within a 16-lane half)
#pragma unroll
            for (int mask = 1; mask < 16; mask <<= 1)
#pragma unroll
                for (int r = 0; r < 8; ++r) p[r] += __shfl_xor(p[r], mask, 32);
            if (ncol == 0) {
#pragma unroll
                for (int r = 0; r < 8; ++r)
                    S.sc[(mbase + r) * 4 + h] = p[r] * 0.25f;  // / sqrt(DK=16)
            }
        }
    }
    for (int i = lane; i < 64; i += 32)
        scoreg[(size_t)e0 * 4 + i] = S.sc[i];
}

// ---------------- segment softmax via atomics ----------------
__global__ void smax_kernel(const float* __restrict__ scoreg, const int* __restrict__ ei,
                            unsigned int* __restrict__ smax_u) {
    int i = blockIdx.x * blockDim.x + threadIdx.x;
    if (i >= NEDGES * 4) return;
    int e = i >> 2, h = i & 3;
    const int* dstI = ei + NEDGES;
    unsigned int b = __float_as_uint(scoreg[i]);
    unsigned int enc = (b & 0x80000000u) ? ~b : (b | 0x80000000u);  // orderable encoding
    atomicMax(&smax_u[dstI[e] * 4 + h], enc);
}

__global__ void esum_kernel(float* __restrict__ scoreg, const int* __restrict__ ei,
                            const unsigned int* __restrict__ smax_u, float* __restrict__ sumexp) {
    int i = blockIdx.x * blockDim.x + threadIdx.x;
    if (i >= NEDGES * 4) return;
    int e = i >> 2, h = i & 3;
    const int* dstI = ei + NEDGES;
    int d = dstI[e];
    unsigned int u = smax_u[d * 4 + h];
    float m = (u & 0x80000000u) ? __uint_as_float(u ^ 0x80000000u) : __uint_as_float(~u);
    float ex = __expf(scoreg[i] - m);
    scoreg[i] = ex;                       // overwrite scores with exp_s
    atomicAdd(&sumexp[d * 4 + h], ex);
}

__global__ void agg_kernel(const float* __restrict__ scoreg, const float* __restrict__ sumexp,
                           const int* __restrict__ ei, const float* __restrict__ Vbuf,
                           float* __restrict__ msg) {
    int i = blockIdx.x * blockDim.x + threadIdx.x;
    if (i >= NEDGES * 64) return;
    int e = i >> 6, cc = i & 63, h = cc >> 4;
    const int* dstI = ei + NEDGES;
    int d = dstI[e];
    float al = scoreg[e * 4 + h] / (sumexp[d * 4 + h] + 1e-12f);
    atomicAdd(&msg[d * 64 + cc], al * Vbuf[i]);
}

// ---------------- node kernel: msg@wo.T, node MLP, LN, residual ----------------
struct NodeSmem {
    __bf16 inA[16 * 128];   // [x | msg@wo.T] bf16
    __bf16 hbf[16 * 64];
    float  hf[16 * 64];
    float  stats[32];
};

__global__ __launch_bounds__(128) void node_kernel(
        const float* __restrict__ x, const float* __restrict__ msg,
        const __bf16* __restrict__ wbf,
        const float* n_b0, const float* n_b1, const float* n_b2,
        const float* n_g, const float* n_be,
        float* __restrict__ x_out) {
    __shared__ NodeSmem smem[4];
    const int lane = threadIdx.x & 31;
    const int wv = threadIdx.x >> 5;
    NodeSmem& S = smem[wv];
    const int n0 = (blockIdx.x * 4 + wv) * 16;
    if (n0 >= NNODES) return;

    const __bf16* w_o  = wbf + 32768;
    const __bf16* w_n0 = wbf + 36864;
    const __bf16* w_n1 = wbf + 45056;
    const __bf16* w_n2 = wbf + 49152;

    const int c = lane * 2;
    for (int r = 0; r < 16; ++r) {
        int n = n0 + r;
        float2 mx = *(const float2*)(x + (size_t)n * 64 + c);
        float2 mm = *(const float2*)(msg + (size_t)n * 64 + c);
        S.inA[r * 128 + c]     = (__bf16)mx.x; S.inA[r * 128 + c + 1] = (__bf16)mx.y;
        S.hbf[r * 64 + c]      = (__bf16)mm.x; S.hbf[r * 64 + c + 1]  = (__bf16)mm.y;
    }

    const int ncol = lane & 15, mbase = (lane >> 4) * 8;

    // msgo = msg @ wo.T -> inA columns 64..127 (bf16)
    {
        v8f oacc[4];
        proj64(S.hbf, w_o, oacc, lane);
#pragma unroll
        for (int t = 0; t < 4; ++t) {
            int n = t * 16 + ncol;
#pragma unroll
            for (int r = 0; r < 8; ++r)
                S.inA[(mbase + r) * 128 + 64 + n] = (__bf16)oacc[t][r];
        }
    }

    // node MLP: 128->64 relu, 64->64 relu, 64->64, LayerNorm
    mlp_layer<128, true >(S.inA, 128, w_n0, n_b0, S.hbf, nullptr, lane);
    mlp_layer<64,  true >(S.hbf,  64, w_n1, n_b1, S.hbf, nullptr, lane);
    mlp_layer<64,  false>(S.hbf,  64, w_n2, n_b2, nullptr, S.hf, lane);
    layer_norm16(S.hf, n_g, n_be, S.stats, lane);

    // residual
    for (int i = lane; i < 16 * 64; i += 32) {
        int r = i >> 6, cc = i & 63;
        size_t n = (size_t)(n0 + r);
        x_out[n * 64 + cc] = x[n * 64 + cc] + S.hf[i];
    }
}

extern "C" void kernel_launch(void* const* d_in, const int* in_sizes, int n_in,
                              void* d_out, int out_size, void* d_ws, size_t ws_size,
                              hipStream_t stream) {
    const float* x     = (const float*)d_in[0];
    const int*   ei    = (const int*)d_in[1];
    const float* eattr = (const float*)d_in[2];
    const float* e_w0 = (const float*)d_in[3],  *e_b0 = (const float*)d_in[4];
    const float* e_w1 = (const float*)d_in[5],  *e_b1 = (const float*)d_in[6];
    const float* e_w2 = (const float*)d_in[7],  *e_b2 = (const float*)d_in[8];
    const float* e_g  = (const float*)d_in[9],  *e_be = (const float*)d_in[10];
    const float* n_w0 = (const float*)d_in[11], *n_b0 = (const float*)d_in[12];
    const float* n_w1 = (const float*)d_in[13], *n_b1 = (const float*)d_in[14];
    const float* n_w2 = (const float*)d_in[15], *n_b2 = (const float*)d_in[16];
    const float* n_g  = (const float*)d_in[17], *n_be = (const float*)d_in[18];
    const float* wq = (const float*)d_in[19], *wk = (const float*)d_in[20];
    const float* wv = (const float*)d_in[21], *wo = (const float*)d_in[22];

    float* out_x = (float*)d_out;                   // [N,64]
    float* out_e = out_x + (size_t)NNODES * 64;     // [E,64]

    // workspace layout: bf16 weights (128KB), then fp32 scratch (~232 MB)
    __bf16* wbf = (__bf16*)d_ws;
    float* fbase   = (float*)((char*)d_ws + 131072);
    float* Vbuf    = fbase;                                   // E*64
    float* scoreg  = Vbuf + (size_t)NEDGES * 64;              // E*4 (scores then exp_s)
    unsigned int* smax_u = (unsigned int*)(scoreg + (size_t)NEDGES * 4);  // N*4
    float* sumexp  = (float*)(smax_u + (size_t)NNODES * 4);   // N*4
    float* msg     = sumexp + (size_t)NNODES * 4;             // N*64

    wconv_kernel<<<(53248 + 255) / 256, 256, 0, stream>>>(
        e_w0, e_w1, e_w2, wq, wk, wv, wo, n_w0, n_w1, n_w2, wbf);
    init_kernel<<<(NNODES * 64 + 255) / 256, 256, 0, stream>>>(smax_u, sumexp, msg);

    edge_kernel<<<NEDGES / 64, 128, 0, stream>>>(
        x, ei, eattr, wbf, e_b0, e_b1, e_b2, e_g, e_be, out_e, Vbuf, scoreg);

    smax_kernel<<<(NEDGES * 4 + 255) / 256, 256, 0, stream>>>(scoreg, ei, smax_u);
    esum_kernel<<<(NEDGES * 4 + 255) / 256, 256, 0, stream>>>(scoreg, ei, smax_u, sumexp);
    agg_kernel<<<(NEDGES * 64 + 255) / 256, 256, 0, stream>>>(scoreg, sumexp, ei, Vbuf, msg);

    node_kernel<<<(NNODES / 16 + 3) / 4, 128, 0, stream>>>(
        x, msg, wbf, n_b0, n_b1, n_b2, n_g, n_be, out_x);
}
